// SimlarityLayer_73254962201099
// MI455X (gfx1250) — matrix operand order repaired and checked by reading, not compiled, not run on hardware
//
#include <hip/hip_runtime.h>

#define B_ 64
#define N_ 512
#define F_ 128

typedef _Float16 h4   __attribute__((ext_vector_type(4)));
typedef _Float16 h8   __attribute__((ext_vector_type(8)));
typedef _Float16 v16h __attribute__((ext_vector_type(16)));
typedef float    v8f  __attribute__((ext_vector_type(8)));

// ---------------------------------------------------------------------------
// Kernel 1: row softmax of weight [N,N] -> P [N,N]
// One 256-thread block per row (each thread owns 2 of the 512 columns).
// ---------------------------------------------------------------------------
__global__ void __launch_bounds__(256) softmax_rows(const float* __restrict__ W,
                                                    float* __restrict__ P) {
    __shared__ float red[256];
    const int row = blockIdx.x;
    const int t   = threadIdx.x;
    const float* w = W + (size_t)row * N_;

    float x0 = w[t];
    float x1 = w[t + 256];

    // max-reduce
    red[t] = fmaxf(x0, x1);
    __syncthreads();
    for (int s = 128; s > 0; s >>= 1) {
        if (t < s) red[t] = fmaxf(red[t], red[t + s]);
        __syncthreads();
    }
    const float m = red[0];
    __syncthreads();

    // exp + sum-reduce
    float e0 = __expf(x0 - m);
    float e1 = __expf(x1 - m);
    red[t] = e0 + e1;
    __syncthreads();
    for (int s = 128; s > 0; s >>= 1) {
        if (t < s) red[t] += red[t + s];
        __syncthreads();
    }
    const float inv = 1.0f / red[0];

    P[(size_t)row * N_ + t]       = e0 * inv;
    P[(size_t)row * N_ + t + 256] = e1 * inv;
}

// ---------------------------------------------------------------------------
// Kernel 2: per (b,i): d[b,i] = rsqrt(1 + sum_j P[i,j]*mask[b,i,j])
// One wave32 per row; 8 rows per 256-thread block. float4 loads.
// ---------------------------------------------------------------------------
__global__ void __launch_bounds__(256) degree_rsqrt(const float* __restrict__ P,
                                                    const float* __restrict__ M,
                                                    float* __restrict__ D) {
    const int wid  = threadIdx.x >> 5;
    const int lane = threadIdx.x & 31;
    const int rg   = blockIdx.x * 8 + wid;   // global row id in [0, B*N)
    const int b    = rg >> 9;                // / N_
    const int i    = rg & (N_ - 1);

    const float4* pm = (const float4*)(M + ((size_t)b * N_ + i) * N_);
    const float4* pp = (const float4*)(P + (size_t)i * N_);

    float s = 0.0f;
#pragma unroll
    for (int j4 = lane; j4 < N_ / 4; j4 += 32) {
        const float4 a = pp[j4];
        const float4 c = pm[j4];
        s += a.x * c.x + a.y * c.y + a.z * c.z + a.w * c.w;
    }
    for (int off = 16; off > 0; off >>= 1)
        s += __shfl_down(s, off, 32);

    if (lane == 0) {
        const float rs = s + 1.0f;  // identity contributes 1 on the diagonal
        D[rg] = (rs > 0.0f) ? rsqrtf(rs) : 0.0f;
    }
}

// ---------------------------------------------------------------------------
// Kernel 3: out[b] = diag(d) * (P.*mask + I) * diag(d) * input[b]
// Block = 256 threads (8 wave32), computes 128 rows x 128 cols per batch.
// Double-buffered fp16 LDS tiles, v_wmma_f32_16x16x32_f16, f32 accumulation.
// ---------------------------------------------------------------------------
__global__ void __launch_bounds__(256)
gcn_norm_gemm(const float* __restrict__ X,   // input [B,N,F]
              const float* __restrict__ Mk,  // mask  [B,N,N]
              const float* __restrict__ P,   // softmax(W) [N,N]
              const float* __restrict__ D,   // d [B,N]
              float* __restrict__ O) {       // out [B,N,F]
    constexpr int LDA  = 40;          // padded row stride in halves (80 B)
    constexpr int TILE = 128 * LDA;   // one LDS tile (10.24 KB)
    __shared__ __align__(16) _Float16 As[2 * TILE];  // A' tiles: 128 rows x 32 k
    __shared__ __align__(16) _Float16 Bs[2 * TILE];  // B^T tiles: 128 f x 32 k
    __shared__ float dk[N_];

    const int b    = blockIdx.y;
    const int i0   = blockIdx.x * 128;
    const int t    = threadIdx.x;
    const int lane = t & 31;
    const int wid  = t >> 5;
    const int mw   = wid & 3;    // row-tile pair: rows [mw*32, mw*32+32)
    const int nw   = wid >> 2;   // col half: cols [nw*64, nw*64+64)

    // preload d[b, :] (used for both B-staging and the final row scaling)
    for (int j = t; j < N_; j += 256) dk[j] = D[(size_t)b * N_ + j];
    __syncthreads();

    const float* Mrow = Mk + ((size_t)b * N_ + i0) * N_;
    const float* Xb   = X + (size_t)b * N_ * F_;

    // Stage one 128x32 A' tile and one 32x128 (transposed) B tile into buffer.
    auto stage = [&](int buf, int k0) {
        _Float16* a   = As + buf * TILE;
        _Float16* bsh = Bs + buf * TILE;
        const bool diag = (k0 + 31 >= i0) && (k0 <= i0 + 127);
        // ---- A' = P .* mask (+ I): 1024 float4 units, 4 per thread ----
#pragma unroll
        for (int j = 0; j < 4; ++j) {
            const int u  = t + 256 * j;
            const int r  = u >> 3;          // row within tile
            const int kc = (u & 7) * 4;     // k quad
            const float4 mv = *(const float4*)&Mrow[(size_t)r * N_ + k0 + kc];
            const float4 pv = *(const float4*)&P[(size_t)(i0 + r) * N_ + k0 + kc];
            float v0 = pv.x * mv.x, v1 = pv.y * mv.y;
            float v2 = pv.z * mv.z, v3 = pv.w * mv.w;
            if (diag) {
                const int dpos = (i0 + r) - (k0 + kc);
                if (dpos == 0) v0 += 1.0f;
                if (dpos == 1) v1 += 1.0f;
                if (dpos == 2) v2 += 1.0f;
                if (dpos == 3) v3 += 1.0f;
            }
            const h4 hv = { (_Float16)v0, (_Float16)v1, (_Float16)v2, (_Float16)v3 };
            *(h4*)&a[r * LDA + kc] = hv;
        }
        // ---- B^T[f][k] = d[k]*input[b,k,f]: 4 k per thread-unit, b64 stores ----
#pragma unroll
        for (int j = 0; j < 4; ++j) {
            const int u  = t + 256 * j;
            const int f  = u & 127;
            const int kb = (u >> 7) * 4;
            h4 hv;
#pragma unroll
            for (int i = 0; i < 4; ++i)
                hv[i] = (_Float16)(Xb[(size_t)(k0 + kb + i) * F_ + f] * dk[k0 + kb + i]);
            *(h4*)&bsh[f * LDA + kb] = hv;
        }
    };

    v8f acc[2][4];
    const v8f vzero = {};
#pragma unroll
    for (int mi = 0; mi < 2; ++mi)
#pragma unroll
        for (int ni = 0; ni < 4; ++ni) acc[mi][ni] = vzero;

    const int hi  = lane >> 4;
    const int rlo = lane & 15;

    stage(0, 0);
    __syncthreads();
    int pb = 0;

    for (int k0 = 0; k0 < N_; k0 += 32) {
        // stage next tile into the other buffer (overlaps with WMMAs below)
        if (k0 + 32 < N_) stage(pb ^ 1, k0 + 32);
        // prefetch mask two tiles ahead (global_prefetch_b8)
        if (k0 + 64 < N_) {
            const float* nxt = &Mrow[(size_t)(t >> 1) * N_ + (k0 + 64) + (t & 1) * 16];
            __builtin_prefetch(nxt, 0, 1);
        }

        const _Float16* a   = As + pb * TILE;
        const _Float16* bsh = Bs + pb * TILE;

        // ---- fragments ----
        v16h afrag[2];
#pragma unroll
        for (int mi = 0; mi < 2; ++mi) {
            const int mt = (mw * 2 + mi) * 16;
            // A 16x32 f16 layout: lane<16 -> row=rlo, K{0..7,16..23};
            //                     lane>=16 -> row=rlo, K{8..15,24..31}
            const h8 a0 = *(const h8*)&a[(mt + rlo) * LDA + hi * 8];
            const h8 a1 = *(const h8*)&a[(mt + rlo) * LDA + 16 + hi * 8];
#pragma unroll
            for (int q = 0; q < 8; ++q) { afrag[mi][q] = a0[q]; afrag[mi][8 + q] = a1[q]; }
        }
        v16h bfrag[4];
#pragma unroll
        for (int ni = 0; ni < 4; ++ni) {
            const int ct = (nw * 4 + ni) * 16;
            // B 32x16 f16 layout: lane<16 -> col=rlo, K{0..7,16..23}; lane>=16 -> K{8..15,24..31}
            const h8 b0 = *(const h8*)&bsh[(ct + rlo) * LDA + hi * 8];
            const h8 b1 = *(const h8*)&bsh[(ct + rlo) * LDA + 16 + hi * 8];
#pragma unroll
            for (int q = 0; q < 8; ++q) { bfrag[ni][q] = b0[q]; bfrag[ni][8 + q] = b1[q]; }
        }
        // ---- 8 WMMAs per wave per k-step ----
#pragma unroll
        for (int mi = 0; mi < 2; ++mi)
#pragma unroll
            for (int ni = 0; ni < 4; ++ni)
                acc[mi][ni] = __builtin_amdgcn_wmma_f32_16x16x32_f16(
                    false, afrag[mi], false, bfrag[ni],
                    (short)0, acc[mi][ni], false, false);

        __syncthreads();
        pb ^= 1;
    }

    // ---- store, scaling rows by d_i (C/D layout: M = v + 8*(lane>=16), N = lane&15)
    const int mbase = (lane >> 4) * 8;
#pragma unroll
    for (int mi = 0; mi < 2; ++mi) {
        const int mt = (mw * 2 + mi) * 16;
#pragma unroll
        for (int ni = 0; ni < 4; ++ni) {
            const int ct = (nw * 4 + ni) * 16;
#pragma unroll
            for (int v = 0; v < 8; ++v) {
                const int row = mt + mbase + v;
                const float val = acc[mi][ni][v] * dk[i0 + row];
                O[((size_t)b * N_ + i0 + row) * F_ + ct + rlo] = val;
            }
        }
    }
}

// ---------------------------------------------------------------------------
extern "C" void kernel_launch(void* const* d_in, const int* in_sizes, int n_in,
                              void* d_out, int out_size, void* d_ws, size_t ws_size,
                              hipStream_t stream) {
    const float* X = (const float*)d_in[0];  // input [B,N,F]
    const float* M = (const float*)d_in[1];  // valid_mask [B,N,N]
    const float* W = (const float*)d_in[2];  // weight [N,N]
    float* O = (float*)d_out;

    float* P = (float*)d_ws;        // N*N floats  (1 MB)
    float* D = P + (size_t)N_ * N_; // B*N floats  (128 KB)

    softmax_rows<<<N_, 256, 0, stream>>>(W, P);
    degree_rsqrt<<<(B_ * N_) / 8, 256, 0, stream>>>(P, M, D);
    dim3 grid(N_ / 128, B_);
    gcn_norm_gemm<<<grid, 256, 0, stream>>>(X, M, P, D, O);
}
